// PhysicalSubspaceBackbone_87101936763281
// MI455X (gfx1250) — matrix operand order repaired and checked by reading, not compiled, not run on hardware
//
#include <hip/hip_runtime.h>
#include <stdint.h>

// ---------------------------------------------------------------------------
// PhysicalSubspaceBackbone for MI455X (gfx1250, wave32, WMMA + async-to-LDS)
//   out = silu(x @ U_base[:, idx]) @ V_base[idx, :]
// bf16 WMMA (v_wmma_f32_16x16x32_bf16), f32 accumulate.
// Global->LDS staging uses GLOBAL_LOAD_ASYNC_TO_LDS_B128 (ASYNCcnt) with
// double-buffered LDS tiles.
// Workspace layout (bytes):
//   [0,       64MB) : Xb  bf16 (8192 x 4096) row-major (M,K)
//   [64MB,    72MB) : Ub  bf16 (1024 x 4096) row-major (N,K)  == U_base[:,idx]^T
//   [72MB,    80MB) : Vb  bf16 (4096 x 1024) row-major (N,K)  == V_base[idx,:]^T
//   [80MB,    96MB) : P   bf16 (8192 x 1024) row-major (M,K)  == silu(x@u)
// ---------------------------------------------------------------------------

#define D_MODEL 4096
#define D_BASE  8192
#define K_SUB   1024
#define N_TOK   8192

typedef __bf16 bf16_t;
typedef __attribute__((ext_vector_type(16))) __bf16 v16bf;
typedef __attribute__((ext_vector_type(8)))  __bf16 v8bf;
typedef __attribute__((ext_vector_type(8)))  float  v8f;

union V16 {
    v16bf v;
    struct { v8bf lo, hi; } h;
};

static __device__ __forceinline__ bf16_t f2bf(float f) {
    uint32_t u = __builtin_bit_cast(uint32_t, f);
    uint32_t r = u + 0x7FFFu + ((u >> 16) & 1u);   // round-to-nearest-even
    unsigned short hs = (unsigned short)(r >> 16);
    return __builtin_bit_cast(bf16_t, hs);
}

// 16-byte async DMA: global -> LDS, tracked by ASYNCcnt (CDNA5 §10, op 98).
// First operand = VGPR holding LDS byte offset, second = 64-bit global addr.
static __device__ __forceinline__ void async_cp16(void* lds_ptr, const void* gptr) {
    uint32_t loff =
        (uint32_t)(uintptr_t)(__attribute__((address_space(3))) void*)lds_ptr;
    asm volatile("global_load_async_to_lds_b128 %0, %1, off"
                 :: "v"(loff), "v"(gptr)
                 : "memory");
}

// ------------------------- precision / gather passes -----------------------

__global__ __launch_bounds__(256)
void k_f32_to_bf16(const float* __restrict__ in, bf16_t* __restrict__ out) {
    int i = (blockIdx.x * 256 + threadIdx.x) * 8;
    const float4* p = (const float4*)(in + i);
    float4 a = p[0];
    float4 b = p[1];
    v8bf o;
    o[0] = f2bf(a.x); o[1] = f2bf(a.y); o[2] = f2bf(a.z); o[3] = f2bf(a.w);
    o[4] = f2bf(b.x); o[5] = f2bf(b.y); o[6] = f2bf(b.z); o[7] = f2bf(b.w);
    *(v8bf*)(out + i) = o;
}

// Ub[n][k] = (bf16) U_base[k][idx[n]]   (column gather -> (N,K) layout)
__global__ __launch_bounds__(256)
void k_gather_u(const float* __restrict__ U, const int* __restrict__ idx,
                bf16_t* __restrict__ Ub) {
    int i = blockIdx.x * 256 + threadIdx.x;      // over D_MODEL * K_SUB
    int k = i >> 10;                             // / K_SUB
    int n = i & (K_SUB - 1);
    float f = U[(size_t)k * D_BASE + idx[n]];
    Ub[(size_t)n * D_MODEL + k] = f2bf(f);
}

// Vb[n][k] = (bf16) V_base[idx[k]][n]   (row gather + transpose -> (N,K))
__global__ __launch_bounds__(256)
void k_gather_v(const float* __restrict__ V, const int* __restrict__ idx,
                bf16_t* __restrict__ Vb) {
    int i = blockIdx.x * 256 + threadIdx.x;      // over K_SUB * D_MODEL
    int k = i >> 12;                             // / D_MODEL
    int n = i & (D_MODEL - 1);
    float f = V[(size_t)idx[k] * D_MODEL + n];   // coalesced along n
    Vb[(size_t)n * K_SUB + k] = f2bf(f);
}

// ------------------------------- WMMA GEMM ---------------------------------
// C(M,N) = [silu] ( A(M,K) * B^T(N,K) ), all bf16 in, f32 accumulate.
// Block tile 128x128x32, 256 threads = 8 waves; each wave: 32(M) x 64(N).
// Double-buffered LDS tiles filled by async DMA; ASYNCcnt-based pipeline.

template <bool SILU, typename OutT>
__global__ __launch_bounds__(256)
void k_gemm_bf16(const bf16_t* __restrict__ A,   // (M,K) row-major
                 const bf16_t* __restrict__ Bt,  // (N,K) row-major
                 OutT* __restrict__ C,           // (M,N) row-major
                 int M, int N, int K) {
    constexpr int BM = 128, BN = 128, BK = 32, LDT = BK + 8;  // 80B row stride
    __shared__ __align__(16) bf16_t sA[2][BM][LDT];
    __shared__ __align__(16) bf16_t sB[2][BN][LDT];

    const int tid  = threadIdx.x;
    const int lane = tid & 31;
    const int wid  = tid >> 5;
    const int wm   = wid & 3;      // wave M index: 0..3  (4*32 = 128)
    const int wn   = wid >> 2;     // wave N index: 0..1  (2*64 = 128)
    const int lm   = lane & 15;    // row/col within 16x16 tile
    const int kh   = lane >> 4;    // K-half select per ISA layout

    const int bm0 = blockIdx.x * BM;
    const int bn0 = blockIdx.y * BN;

    // Per-thread staging chunks: 512 x 16B chunks per tile, 2 per thread.
    const int r0 = tid >> 2,            kc0 = (tid & 3) * 8;
    const int r1 = (tid + 256) >> 2,    kc1 = ((tid + 256) & 3) * 8;
    const bf16_t* gA0 = A  + (size_t)(bm0 + r0) * K + kc0;
    const bf16_t* gA1 = A  + (size_t)(bm0 + r1) * K + kc1;
    const bf16_t* gB0 = Bt + (size_t)(bn0 + r0) * K + kc0;
    const bf16_t* gB1 = Bt + (size_t)(bn0 + r1) * K + kc1;

    v8f acc[2][4];
    v8f zero = {};
#pragma unroll
    for (int im = 0; im < 2; ++im)
#pragma unroll
        for (int in = 0; in < 4; ++in)
            acc[im][in] = zero;

    const int KT = K / BK;

    // Prologue: async-fill buffer 0 with tile 0 (4 DMAs per thread).
    async_cp16(&sA[0][r0][kc0], gA0);
    async_cp16(&sA[0][r1][kc1], gA1);
    async_cp16(&sB[0][r0][kc0], gB0);
    async_cp16(&sB[0][r1][kc1], gB1);

    for (int kt = 0; kt < KT; ++kt) {
        const int buf  = kt & 1;
        const bool more = (kt + 1) < KT;

        if (more) {
            const int nb   = buf ^ 1;
            const int koff = (kt + 1) * BK;
            async_cp16(&sA[nb][r0][kc0], gA0 + koff);
            async_cp16(&sA[nb][r1][kc1], gA1 + koff);
            async_cp16(&sB[nb][r0][kc0], gB0 + koff);
            async_cp16(&sB[nb][r1][kc1], gB1 + koff);
            // This wave's tile-kt DMAs (first 4 of 8 outstanding) complete
            // in order -> wait ASYNCcnt <= 4.
            asm volatile("s_wait_asynccnt 0x4" ::: "memory");
        } else {
            asm volatile("s_wait_asynccnt 0x0" ::: "memory");
        }
        __syncthreads();   // all waves' tile-kt DMAs visible in LDS

        // ---- LDS -> fragments (ISA 7.12.2 16-bit layouts) ----
        V16 afrag[2], bfrag[4];
#pragma unroll
        for (int im = 0; im < 2; ++im) {
            int r = wm * 32 + im * 16 + lm;          // A row M
            afrag[im].h.lo = *(const v8bf*)&sA[buf][r][kh * 8];      // K 0-7 / 8-15
            afrag[im].h.hi = *(const v8bf*)&sA[buf][r][16 + kh * 8]; // K 16-23 / 24-31
        }
#pragma unroll
        for (int in = 0; in < 4; ++in) {
            int r = wn * 64 + in * 16 + lm;          // B column N
            bfrag[in].h.lo = *(const v8bf*)&sB[buf][r][kh * 8];      // K 0-7 / 8-15
            bfrag[in].h.hi = *(const v8bf*)&sB[buf][r][16 + kh * 8]; // K 16-23 / 24-31
        }

        // ---- 8x v_wmma_f32_16x16x32_bf16 per wave per k-step ----
#pragma unroll
        for (int im = 0; im < 2; ++im)
#pragma unroll
            for (int in = 0; in < 4; ++in)
                acc[im][in] = __builtin_amdgcn_wmma_f32_16x16x32_bf16(
                    false, afrag[im].v, false, bfrag[in].v,
                    (short)0, acc[im][in], false, false);

        __syncthreads();   // protect buf from next iteration's DMA overwrite
    }

    // ---- epilogue: C layout (VGPR r: lanes0-15 M=r, lanes16-31 M=8+r) ----
#pragma unroll
    for (int im = 0; im < 2; ++im) {
#pragma unroll
        for (int in = 0; in < 4; ++in) {
            int m0  = bm0 + wm * 32 + im * 16 + kh * 8;
            int col = bn0 + wn * 64 + in * 16 + lm;
#pragma unroll
            for (int r = 0; r < 8; ++r) {
                float v = acc[im][in][r];
                if (SILU) v = v / (1.0f + __expf(-v));
                size_t o = (size_t)(m0 + r) * N + col;
                if constexpr (sizeof(OutT) == 2)
                    ((bf16_t*)C)[o] = f2bf(v);
                else
                    ((float*)C)[o] = v;
            }
        }
    }
}

// ------------------------------- launcher ----------------------------------

extern "C" void kernel_launch(void* const* d_in, const int* in_sizes, int n_in,
                              void* d_out, int out_size, void* d_ws, size_t ws_size,
                              hipStream_t stream) {
    const float* x   = (const float*)d_in[0];
    const int*   idx = (const int*)d_in[1];
    const float* U   = (const float*)d_in[2];
    const float* V   = (const float*)d_in[3];
    float* out = (float*)d_out;

    char* ws = (char*)d_ws;
    const size_t szX = (size_t)N_TOK * D_MODEL * 2;   // 64 MB
    const size_t szU = (size_t)K_SUB * D_MODEL * 2;   //  8 MB
    const size_t szV = (size_t)K_SUB * D_MODEL * 2;   //  8 MB
    bf16_t* Xb = (bf16_t*)(ws);
    bf16_t* Ub = (bf16_t*)(ws + szX);
    bf16_t* Vb = (bf16_t*)(ws + szX + szU);
    bf16_t* P  = (bf16_t*)(ws + szX + szU + szV);     // 16 MB

    // 1) x -> bf16  (32M elems, 8 per thread)
    k_f32_to_bf16<<<(N_TOK * D_MODEL) / (256 * 8), 256, 0, stream>>>(x, Xb);
    // 2) gather U columns -> (N,K) bf16
    k_gather_u<<<(D_MODEL * K_SUB) / 256, 256, 0, stream>>>(U, idx, Ub);
    // 3) gather V rows -> (N,K) bf16
    k_gather_v<<<(D_MODEL * K_SUB) / 256, 256, 0, stream>>>(V, idx, Vb);

    dim3 blk(256);
    // 4) P = silu(Xb @ Ub^T)    M=8192, N=1024, K=4096
    dim3 g1(N_TOK / 128, K_SUB / 128);
    k_gemm_bf16<true, bf16_t><<<g1, blk, 0, stream>>>(Xb, Ub, P, N_TOK, K_SUB, D_MODEL);
    // 5) out = P @ Vb^T         M=8192, N=4096, K=1024
    dim3 g2(N_TOK / 128, D_MODEL / 128);
    k_gemm_bf16<false, float><<<g2, blk, 0, stream>>>(P, Vb, out, N_TOK, D_MODEL, K_SUB);
}
